// Generator_21603685499469
// MI455X (gfx1250) — compile-verified, matrix operand order checked by reading, not run on hardware
//
#include <hip/hip_runtime.h>

typedef __attribute__((ext_vector_type(16))) __bf16 v16bf;
typedef __attribute__((ext_vector_type(8)))  __bf16 v8bf;
typedef __attribute__((ext_vector_type(8)))  float  v8f;

#define B_   32
#define T_   1024
#define E_   64
#define H0_  512

union Frag16 { v16bf v; v8bf h[2]; };

// ---------------------------------------------------------------------------
// CDNA5 async global->LDS copy (16B per lane), tracked by ASYNCcnt.
// LDS destination address comes from the vdst VGPR; generic LDS pointer low 32 bits are the LDS
// byte offset (ISA 10.2 aperture).
// ---------------------------------------------------------------------------
__device__ __forceinline__ void async_g2l_b128(void* lds, const void* gptr)
{
    asm volatile("global_load_async_to_lds_b128 %0, %1, off"
                 :: "v"((unsigned)(size_t)lds),
                    "v"((unsigned long long)(size_t)gptr)
                 : "memory");
}
__device__ __forceinline__ void wait_async0()
{
    asm volatile("s_wait_asynccnt 0x0" ::: "memory");
}

// ---------------------------------------------------------------------------
// WMMA GEMM (NT only): C[M,N] = act(scale * (A @ Bt^T) + bias)
//   A  : bf16 [M,K] row-major, lda
//   Bt : bf16 [N,K] row-major, ldb   (i.e. B already transposed)
//   C  : fp32 or bf16 [M,N], ldc
// 128 threads = 4 waves. Block tile 128(M) x 128(N); per-wave 32x128
// (2 A-frags x 8 B-frags = 16 v_wmma per K-step of 32).
// B tile (128x32, 8KB) staged into LDS via GLOBAL_LOAD_ASYNC_TO_LDS_B128 and
// shared by all 4 waves. Requires N%128==0, K%32==0 (true for this net);
// M handled by per-wave active flag (waves stay in barriers, EXEC all-ones).
// ---------------------------------------------------------------------------
#define BSTRIDE 40   // LDS row stride in elems (80B): conflict-spreading, 16B aligned

template<bool BIAS, bool RELU, bool OUT_BF16>
__global__ __launch_bounds__(128)
void wmma_gemm(const __bf16* __restrict__ A, int lda,
               const __bf16* __restrict__ Bt, int ldb,
               const float* __restrict__ bias,
               void* __restrict__ Cv, int ldc,
               int M, int N, int K, float scale)
{
    __shared__ __align__(16) __bf16 Bs[128][BSTRIDE];

    const int tid  = threadIdx.x;
    const int lane = tid & 31;
    const int wave = tid >> 5;
    const int lrow = lane & 15;   // row (A) / col (B) within 16x16 tile
    const int lhi  = lane >> 4;   // K sub-block select

    const int n0 = blockIdx.x * 128;
    const int m0 = blockIdx.y * 128 + wave * 32;
    const bool active = (m0 < M);
    const int mr0 = active ? (m0 + lrow)      : 0;  // clamp: no OOB loads
    const int mr1 = active ? (m0 + 16 + lrow) : 0;

    v8f acc[2][8] = {};

    const __bf16* brow = Bt + (size_t)(n0 + tid) * ldb;  // this thread stages row n0+tid
    __bf16* dst = &Bs[tid][0];

    for (int k0 = 0; k0 < K; k0 += 32) {
        // ---- async-stage B tile: rows n0..n0+127, cols k0..k0+31 (64B/row)
        const __bf16* src = brow + k0;
        async_g2l_b128(dst,      src);
        async_g2l_b128(dst + 8,  src + 8);
        async_g2l_b128(dst + 16, src + 16);
        async_g2l_b128(dst + 24, src + 24);
        wait_async0();
        __syncthreads();

        // ---- A fragments straight from global (16B vector loads)
        Frag16 a[2];
        const __bf16* ap0 = A + (size_t)mr0 * lda + k0 + lhi * 8;
        const __bf16* ap1 = A + (size_t)mr1 * lda + k0 + lhi * 8;
        a[0].h[0] = *(const v8bf*)ap0;  a[0].h[1] = *(const v8bf*)(ap0 + 16);
        a[1].h[0] = *(const v8bf*)ap1;  a[1].h[1] = *(const v8bf*)(ap1 + 16);

        // ---- 8 B fragments from LDS, 16 WMMAs
        #pragma unroll
        for (int t = 0; t < 8; ++t) {
            Frag16 b;
            const __bf16* bp = &Bs[t * 16 + lrow][lhi * 8];
            b.h[0] = *(const v8bf*)bp;
            b.h[1] = *(const v8bf*)(bp + 16);
            acc[0][t] = __builtin_amdgcn_wmma_f32_16x16x32_bf16(
                false, a[0].v, false, b.v, (short)0, acc[0][t], false, false);
            acc[1][t] = __builtin_amdgcn_wmma_f32_16x16x32_bf16(
                false, a[1].v, false, b.v, (short)0, acc[1][t], false, false);
        }
        __syncthreads();   // protect Bs before next stage
    }

    if (!active) return;

    // ---- epilogue: C/D layout -> row = m0 + 16*i + 8*lhi + j, col = n0+16t+lrow
    #pragma unroll
    for (int t = 0; t < 8; ++t) {
        const int col = n0 + t * 16 + lrow;
        const float bcol = BIAS ? bias[col] : 0.0f;
        #pragma unroll
        for (int i = 0; i < 2; ++i) {
            #pragma unroll
            for (int j = 0; j < 8; ++j) {
                const int row = m0 + i * 16 + lhi * 8 + j;
                float v = acc[i][t][j] * scale + bcol;
                if (RELU) v = (v >= 0.0f) ? v : 0.01f * v;   // LeakyReLU(0.01)
                if (OUT_BF16) ((__bf16*)Cv)[(size_t)row * ldc + col] = (__bf16)v;
                else          ((float*) Cv)[(size_t)row * ldc + col] = v;
            }
        }
    }
}

// ---------------------------------------------------------------------------
// z[b,t,h] = y[b,t]*pw[0,h] + sum_e g[b,e]*pw[1+e,h] + pb[h] + pos_enc(t,h)
// ---------------------------------------------------------------------------
__global__ void embed_proj_kernel(const float* __restrict__ y,
                                  const int* __restrict__ group,
                                  const float* __restrict__ etab,
                                  const float* __restrict__ pw,
                                  const float* __restrict__ pb,
                                  float* __restrict__ z)
{
    const int bt = blockIdx.x;
    const int b  = bt / T_;
    const int t  = bt % T_;
    const float yv = y[bt];
    const float* g = etab + (size_t)group[b] * E_;
    for (int h = threadIdx.x; h < H0_; h += blockDim.x) {
        float acc = pb[h] + yv * pw[h];
        #pragma unroll 8
        for (int e = 0; e < E_; ++e)
            acc += g[e] * pw[(size_t)(1 + e) * H0_ + h];
        const int   i   = h & ~1;
        const float div = __expf(-(float)i * (9.210340371976184f / (float)H0_));
        const float pe  = (h & 1) ? __cosf((float)t * div) : __sinf((float)t * div);
        z[(size_t)bt * H0_ + h] = acc + pe;
    }
}

// ---------------------------------------------------------------------------
// LayerNorm over D=512, fp32 in -> bf16 out. One 128-thread block per row.
// ---------------------------------------------------------------------------
__global__ __launch_bounds__(128)
void ln_kernel(const float* __restrict__ x, const float* __restrict__ gamma,
               const float* __restrict__ beta, __bf16* __restrict__ out, int D)
{
    __shared__ float s1[128];
    __shared__ float s2[128];
    const int row = blockIdx.x;
    const int tid = threadIdx.x;
    const float* xr = x + (size_t)row * D;
    float a = 0.0f, b = 0.0f;
    for (int i = tid; i < D; i += 128) { float v = xr[i]; a += v; b += v * v; }
    s1[tid] = a; s2[tid] = b; __syncthreads();
    for (int s = 64; s > 0; s >>= 1) {
        if (tid < s) { s1[tid] += s1[tid + s]; s2[tid] += s2[tid + s]; }
        __syncthreads();
    }
    const float mu  = s1[0] / (float)D;
    const float var = s2[0] / (float)D - mu * mu;
    const float inv = rsqrtf(var + 1e-5f);
    for (int i = tid; i < D; i += 128) {
        float v = (xr[i] - mu) * inv * gamma[i] + beta[i];
        out[(size_t)row * D + i] = (__bf16)v;
    }
}

// ---------------------------------------------------------------------------
// Row softmax: S fp32 [rows,T] -> P bf16. One 256-thread block per row.
// ---------------------------------------------------------------------------
__global__ __launch_bounds__(256)
void softmax_kernel(const float* __restrict__ S, __bf16* __restrict__ P, int T)
{
    __shared__ float red[256];
    const int row = blockIdx.x;
    const int tid = threadIdx.x;
    const float* sr = S + (size_t)row * T;

    float m = -3.402823466e38f;
    for (int i = tid; i < T; i += 256) m = fmaxf(m, sr[i]);
    red[tid] = m; __syncthreads();
    for (int s = 128; s > 0; s >>= 1) {
        if (tid < s) red[tid] = fmaxf(red[tid], red[tid + s]);
        __syncthreads();
    }
    const float rowmax = red[0];
    __syncthreads();

    float sum = 0.0f;
    for (int i = tid; i < T; i += 256) sum += __expf(sr[i] - rowmax);
    red[tid] = sum; __syncthreads();
    for (int s = 128; s > 0; s >>= 1) {
        if (tid < s) red[tid] += red[tid + s];
        __syncthreads();
    }
    const float inv = 1.0f / red[0];
    for (int i = tid; i < T; i += 256)
        P[(size_t)row * T + i] = (__bf16)(__expf(sr[i] - rowmax) * inv);
}

// ---------------------------------------------------------------------------
// Encoder-3 FC (dout=1) + LeakyReLU
// ---------------------------------------------------------------------------
__global__ void fc_dout1_kernel(const __bf16* __restrict__ o,
                                const float* __restrict__ w,
                                const float* __restrict__ bias,
                                float* __restrict__ out, int R, int D)
{
    const int r = blockIdx.x * blockDim.x + threadIdx.x;
    if (r >= R) return;
    const __bf16* orow = o + (size_t)r * D;
    float acc = bias[0];
    for (int i = 0; i < D; ++i) acc += (float)orow[i] * w[i];
    out[r] = (acc >= 0.0f) ? acc : 0.01f * acc;
}

// fp32 [K,N] -> bf16 transposed [N,K] (so all GEMMs use the NT fast path)
__global__ void cvt_bf16_t_kernel(const float* __restrict__ in,
                                  __bf16* __restrict__ out, int K, int N)
{
    const int idx = blockIdx.x * blockDim.x + threadIdx.x;
    if (idx >= K * N) return;
    const int k = idx / N, n = idx % N;
    out[(size_t)n * K + k] = (__bf16)in[idx];
}

__global__ void cvt_bf16_kernel(const float* __restrict__ in,
                                __bf16* __restrict__ out, int n)
{
    const int i = blockIdx.x * blockDim.x + threadIdx.x;
    if (i < n) out[i] = (__bf16)in[i];
}

// V slice of qkv [B,T,3H] -> Vt [B, H0, T] (bf16), so PV GEMM is NT too
__global__ void transpose_v_kernel(const __bf16* __restrict__ qkv,
                                   __bf16* __restrict__ vt)
{
    const int idx = blockIdx.x * blockDim.x + threadIdx.x;
    if (idx >= B_ * T_ * H0_) return;
    const int d = idx & (H0_ - 1);
    const int t = (idx >> 9) & (T_ - 1);
    const int b = idx >> 19;
    vt[((size_t)b * H0_ + d) * T_ + t] =
        qkv[(size_t)(b * T_ + t) * (3 * H0_) + 2 * H0_ + d];
}

// ---------------------------------------------------------------------------
extern "C" void kernel_launch(void* const* d_in, const int* in_sizes, int n_in,
                              void* d_out, int out_size, void* d_ws, size_t ws_size,
                              hipStream_t stream)
{
    (void)in_sizes; (void)n_in; (void)out_size; (void)ws_size;
    const float* y     = (const float*)d_in[0];
    const int*   group = (const int*)  d_in[1];
    const float* etab  = (const float*)d_in[2];
    const float* projw = (const float*)d_in[3];
    const float* projb = (const float*)d_in[4];
    const float* finw  = (const float*)d_in[25];
    const float* finb  = (const float*)d_in[26];

    char* ws = (char*)d_ws;
    size_t off = 0;
    auto carve = [&](size_t bytes) {
        char* p = ws + off;
        off += (bytes + 255) & ~(size_t)255;
        return p;
    };
    float*  z     = (float*) carve((size_t)B_ * T_ * H0_ * 4);      // 64 MiB
    __bf16* xn    = (__bf16*)carve((size_t)B_ * T_ * H0_ * 2);      // 32 MiB
    __bf16* qkv   = (__bf16*)carve((size_t)B_ * T_ * 3 * H0_ * 2);  // 96 MiB
    __bf16* vt    = (__bf16*)carve((size_t)B_ * H0_ * T_ * 2);      // 32 MiB
    __bf16* ob    = (__bf16*)carve((size_t)B_ * T_ * H0_ * 2);      // 32 MiB
    float*  S     = (float*) carve((size_t)T_ * T_ * 4);            //  4 MiB
    __bf16* P     = (__bf16*)carve((size_t)T_ * T_ * 2);            //  2 MiB
    __bf16* wqT   = (__bf16*)carve((size_t)H0_ * 3 * H0_ * 2);      // [1536,512]
    __bf16* wfT   = (__bf16*)carve((size_t)H0_ * H0_ * 2);          // [512,512]
    __bf16* wfinT = (__bf16*)carve((size_t)T_ * T_ * 2);            // [1024,1024]
    float*  zsq   = (float*) carve((size_t)B_ * T_ * 4);
    __bf16* zsqb  = (__bf16*)carve((size_t)B_ * T_ * 2);

    embed_proj_kernel<<<dim3(B_ * T_), 256, 0, stream>>>(y, group, etab, projw, projb, z);

    const float attn_scale = 0.044194173824159216f; // 512^-0.5

    for (int e = 0; e < 4; ++e) {
        const float* gamma = (const float*)d_in[5 + e * 5 + 0];
        const float* beta  = (const float*)d_in[5 + e * 5 + 1];
        const float* qkvw  = (const float*)d_in[5 + e * 5 + 2];
        const float* fcw   = (const float*)d_in[5 + e * 5 + 3];
        const float* fcb   = (const float*)d_in[5 + e * 5 + 4];

        cvt_bf16_t_kernel<<<(H0_ * 3 * H0_ + 255) / 256, 256, 0, stream>>>(
            qkvw, wqT, H0_, 3 * H0_);
        ln_kernel<<<B_ * T_, 128, 0, stream>>>(z, gamma, beta, xn, H0_);

        // qkv = xn @ qkv_w  (M=32768, N=1536, K=512)
        wmma_gemm<false, false, true>
            <<<dim3((3 * H0_) / 128, (B_ * T_) / 128), 128, 0, stream>>>(
                xn, H0_, wqT, H0_, nullptr, qkv, 3 * H0_,
                B_ * T_, 3 * H0_, H0_, 1.0f);

        transpose_v_kernel<<<(B_ * T_ * H0_ + 255) / 256, 256, 0, stream>>>(qkv, vt);

        for (int b = 0; b < B_; ++b) {
            const __bf16* qb = qkv + (size_t)b * T_ * (3 * H0_);
            const __bf16* kb = qb + H0_;
            // S = scale * q @ k^T  (M=N=1024, K=512)
            wmma_gemm<false, false, false>
                <<<dim3(T_ / 128, T_ / 128), 128, 0, stream>>>(
                    qb, 3 * H0_, kb, 3 * H0_, nullptr, S, T_,
                    T_, T_, H0_, attn_scale);
            softmax_kernel<<<T_, 256, 0, stream>>>(S, P, T_);
            // O = P @ V  (M=1024, N=512, K=1024), Bt = Vt[b] is [512, T]
            wmma_gemm<false, false, true>
                <<<dim3(H0_ / 128, T_ / 128), 128, 0, stream>>>(
                    P, T_, vt + (size_t)b * H0_ * T_, T_, nullptr,
                    ob + (size_t)b * T_ * H0_, H0_,
                    T_, H0_, T_, 1.0f);
        }

        if (e < 3) {
            cvt_bf16_t_kernel<<<(H0_ * H0_ + 255) / 256, 256, 0, stream>>>(
                fcw, wfT, H0_, H0_);
            // z = LeakyReLU(O @ fc_w + fc_b)  (M=32768, N=512, K=512)
            wmma_gemm<true, true, false>
                <<<dim3(H0_ / 128, (B_ * T_) / 128), 128, 0, stream>>>(
                    ob, H0_, wfT, H0_, fcb, z, H0_,
                    B_ * T_, H0_, H0_, 1.0f);
        } else {
            fc_dout1_kernel<<<(B_ * T_ + 255) / 256, 256, 0, stream>>>(
                ob, fcw, fcb, zsq, B_ * T_, H0_);
        }
    }

    // final: out = zsq @ fc_w + fc_b  (M=32, N=1024, K=1024)
    cvt_bf16_kernel<<<(B_ * T_ + 255) / 256, 256, 0, stream>>>(zsq, zsqb, B_ * T_);
    cvt_bf16_t_kernel<<<(T_ * T_ + 255) / 256, 256, 0, stream>>>(finw, wfinT, T_, T_);
    wmma_gemm<true, false, false>
        <<<dim3(T_ / 128, 1), 128, 0, stream>>>(
            zsqb, T_, wfinT, T_, finb, (float*)d_out, T_,
            B_, T_, T_, 1.0f);
}
